// GNN_63702954934563
// MI455X (gfx1250) — matrix operand order repaired and checked by reading, not compiled, not run on hardware
//
#include <hip/hip_runtime.h>
#include <hip/hip_bf16.h>

typedef _Float16 v16h __attribute__((ext_vector_type(16)));
typedef _Float16 v8h  __attribute__((ext_vector_type(8)));
typedef _Float16 h2v  __attribute__((ext_vector_type(2)));
typedef float    v8f  __attribute__((ext_vector_type(8)));

#define BB    4
#define NN    4096
#define DD    64
#define TIMD5 5
#define ALPHA 0.2f
#define L2E   1.44269504f

// ---------------------------------------------------------------------------
// Kernel 1: per (b,n): xp = x@W_w + W_b  -> xpT[b][d][n] (f16, transposed)
//           si = [xp,emb]@a1 + A_b (folded),  sj = [xp,emb]@a2
// ---------------------------------------------------------------------------
__global__ void gat_prep(const float* __restrict__ x, const float* __restrict__ emb,
                         const float* __restrict__ Ww, const float* __restrict__ Wb,
                         const float* __restrict__ Aw, const float* __restrict__ Ab,
                         _Float16* __restrict__ xpT, float* __restrict__ si,
                         float* __restrict__ sj)
{
    int g = blockIdx.x * blockDim.x + threadIdx.x;   // 0 .. B*N-1
    int b = g >> 12;
    int n = g & (NN - 1);

    float xv[TIMD5];
#pragma unroll
    for (int t = 0; t < TIMD5; ++t) xv[t] = x[(size_t)g * TIMD5 + t];

    float s1 = Ab[0];   // fold A_b into si
    float s2 = 0.f;
    _Float16* xo = xpT + (size_t)b * DD * NN + n;

#pragma unroll
    for (int d = 0; d < DD; ++d) {
        float s = Wb[d];
#pragma unroll
        for (int t = 0; t < TIMD5; ++t) s = fmaf(xv[t], Ww[t * DD + d], s);
        xo[(size_t)d * NN] = (_Float16)s;            // coalesced across n
        s1 = fmaf(s, Aw[d],          s1);            // a1[:64]
        s2 = fmaf(s, Aw[2 * DD + d], s2);            // a2[:64]
    }
    const float* er = emb + (size_t)n * DD;
#pragma unroll
    for (int d = 0; d < DD; ++d) {
        float e = er[d];
        s1 = fmaf(e, Aw[DD + d],     s1);            // a1[64:]
        s2 = fmaf(e, Aw[3 * DD + d], s2);            // a2[64:]
    }
    si[g] = s1;
    sj[g] = s2;
}

// ---------------------------------------------------------------------------
// Kernel 2: sjmax[b] = max_j sj[b][j]; then precompute the factored softmax
// column tables:  F = exp(sj - sjmax) <= 1,  G = exp(0.2*(sj - sjmax)) <= 1
// ---------------------------------------------------------------------------
__global__ void gat_sjmax_fg(const float* __restrict__ sj, float* __restrict__ sjmax,
                             float2* __restrict__ FG)
{
    __shared__ float red[256];
    int b = blockIdx.x;
    float m = -3.402823466e38f;
    for (int j = threadIdx.x; j < NN; j += 256) m = fmaxf(m, sj[b * NN + j]);
    red[threadIdx.x] = m;
    __syncthreads();
    for (int s = 128; s > 0; s >>= 1) {
        if ((int)threadIdx.x < s) red[threadIdx.x] = fmaxf(red[threadIdx.x], red[threadIdx.x + s]);
        __syncthreads();
    }
    m = red[0];
    if (threadIdx.x == 0) sjmax[b] = m;
    for (int j = threadIdx.x; j < NN; j += 256) {
        float d = sj[b * NN + j] - m;
        FG[b * NN + j] = make_float2(exp2f(d * L2E), exp2f(d * (ALPHA * L2E)));
    }
}

// ---------------------------------------------------------------------------
// Kernel 3: one wave per (b, 16-row i-tile). Rank-1 factored softmax:
//   tt = si'+sjmax;  Ai=exp(min(tt,0)), Bi=exp(0.2tt-max(tt,0)), Ti=exp(-tt)
//   p = mask ? (Fj > Ti ? Ai*Fj : Bi*Gj) : 0        (all factors <= 1)
//   acc[16x64] += P(16x32,f16) @ xpT(32x64,f16) via 4x v_wmma_f32_16x16x32_f16
//   out = relu(acc / rowsum)     -- no transcendentals, no branches in loop
// ---------------------------------------------------------------------------
__global__ __launch_bounds__(32) void gat_attn(
    const int* __restrict__ mask, const float* __restrict__ si,
    const float* __restrict__ sjmax, const float2* __restrict__ FG,
    const _Float16* __restrict__ xpT, float* __restrict__ out)
{
    __shared__ __align__(16) float4   sh_row[16];     // (Ai, Bi, Ti, -) per row
    __shared__ __align__(16) _Float16 shP[16 * 32];   // P tile, row-major
    __shared__ float sh_l[16];

    const int lane = threadIdx.x;     // 0..31
    const int hf   = lane >> 4;       // half-wave id
    const int c    = lane & 15;
    const int b    = blockIdx.x >> 8; // 256 i-tiles per batch
    const int i0   = (blockIdx.x & 255) << 4;

    if (lane < 16) {
        float tt = si[b * NN + i0 + lane] + sjmax[b];
        float m  = fmaxf(tt, 0.f);
        float Ai = exp2f(fminf(tt, 0.f) * L2E);
        float Bi = exp2f((ALPHA * tt - m) * L2E);
        float Ti = exp2f(-tt * L2E);              // Fj > Ti  <=>  si'+sj > 0
        sh_row[lane] = make_float4(Ai, Bi, Ti, 0.f);
    }
    __syncthreads();

    float4 srows[8];
#pragma unroll
    for (int rr = 0; rr < 8; ++rr) srows[rr] = sh_row[2 * rr + hf];

    v8f acc0 = {}, acc1 = {}, acc2 = {}, acc3 = {};
    float lp0 = 0.f, lp1 = 0.f, lp2 = 0.f, lp3 = 0.f;   // independent chains
    const _Float16* xb  = xpT + (size_t)b * DD * NN;
    const float2*   FGb = FG + (size_t)b * NN;

    for (int jb = 0; jb < NN; jb += 32) {
        // {F0,G0,F1,G1} for this lane's column pair
        float4 fg = *(const float4*)(FGb + jb + 2 * c);

        // ---- compute P tile: lane covers column pair (2c,2c+1), rows 2rr+hf
#pragma unroll
        for (int rr = 0; rr < 8; ++rr) {
            int row = 2 * rr + hf;
            int2 mk = *(const int2*)(mask + (size_t)(i0 + row) * NN + jb + 2 * c);
            float4 s4 = srows[rr];
            float a0 = fg.x > s4.z ? s4.x : s4.y;   // branchless lrelu split
            float f0 = fg.x > s4.z ? fg.x : fg.y;
            float a1 = fg.z > s4.z ? s4.x : s4.y;
            float f1 = fg.z > s4.z ? fg.z : fg.w;
            float p0 = mk.x > 0 ? a0 * f0 : 0.f;
            float p1 = mk.y > 0 ? a1 * f1 : 0.f;
            h2v ph; ph.x = (_Float16)p0; ph.y = (_Float16)p1;
            *(h2v*)(shP + row * 32 + 2 * c) = ph;
        }
        __syncthreads();

        // ---- A fragment (16x32 f16): lane row = c; lanes<16: K{0-15},
        //      lanes>=16: K{16-31}
        v8h lo = *(const v8h*)(shP + c * 32 + hf * 16);
        v8h hi = *(const v8h*)(shP + c * 32 + hf * 16 + 8);
        v16h a = __builtin_shufflevector(lo, hi, 0, 1, 2, 3, 4, 5, 6, 7,
                                                 8, 9, 10, 11, 12, 13, 14, 15);
        // row-sum partials: 4 independent chains (fill WMMA co-exec slots)
#pragma unroll
        for (int k = 0; k < 4; ++k) {
            lp0 += (float)a[k];
            lp1 += (float)a[k + 4];
            lp2 += (float)a[k + 8];
            lp3 += (float)a[k + 12];
        }

        // ---- B fragments from xpT: column d = 16t + c, K rows = jb + hf*16 ..
        const _Float16* bp = xb + jb + hf * 16;
        v16h b0 = *(const v16h*)(bp + (size_t)(c +  0) * NN);
        v16h b1 = *(const v16h*)(bp + (size_t)(c + 16) * NN);
        v16h b2 = *(const v16h*)(bp + (size_t)(c + 32) * NN);
        v16h b3 = *(const v16h*)(bp + (size_t)(c + 48) * NN);

        acc0 = __builtin_amdgcn_wmma_f32_16x16x32_f16(false, a, false, b0, (short)0, acc0, false, false);
        acc1 = __builtin_amdgcn_wmma_f32_16x16x32_f16(false, a, false, b1, (short)0, acc1, false, false);
        acc2 = __builtin_amdgcn_wmma_f32_16x16x32_f16(false, a, false, b2, (short)0, acc2, false, false);
        acc3 = __builtin_amdgcn_wmma_f32_16x16x32_f16(false, a, false, b3, (short)0, acc3, false, false);

        __syncthreads();   // protect shP before next tile's stores
    }

    // ---- full row sums: lane and lane^16 hold the two halves of row c
    float lpart = (lp0 + lp1) + (lp2 + lp3);
    float lrow  = lpart + __shfl_xor(lpart, 16, 32);
    if (lane < 16) sh_l[c] = lrow;
    __syncthreads();

    // ---- normalize + relu + store (C layout: VGPR v -> row v + 8*hf, col c)
#pragma unroll
    for (int v = 0; v < 8; ++v) {
        int row = v + hf * 8;
        float inv = 1.f / sh_l[row];
        float* orow = out + ((size_t)b * NN + i0 + row) * DD + c;
        float z0 = acc0[v] * inv; orow[0]  = z0 > 0.f ? z0 : 0.f;
        float z1 = acc1[v] * inv; orow[16] = z1 > 0.f ? z1 : 0.f;
        float z2 = acc2[v] * inv; orow[32] = z2 > 0.f ? z2 : 0.f;
        float z3 = acc3[v] * inv; orow[48] = z3 > 0.f ? z3 : 0.f;
    }
}

// ---------------------------------------------------------------------------
extern "C" void kernel_launch(void* const* d_in, const int* in_sizes, int n_in,
                              void* d_out, int out_size, void* d_ws, size_t ws_size,
                              hipStream_t stream)
{
    const float* x   = (const float*)d_in[0];   // (B,N,5)
    const float* emb = (const float*)d_in[1];   // (N,64)
    const int*   msk = (const int*)  d_in[2];   // (N,N)
    const float* Ww  = (const float*)d_in[3];   // (5,64)
    const float* Wb  = (const float*)d_in[4];   // (64,)
    const float* Aw  = (const float*)d_in[5];   // (256,)
    const float* Ab  = (const float*)d_in[6];   // scalar
    float* outp = (float*)d_out;                // (B,N,64)

    char* ws = (char*)d_ws;
    _Float16* xpT = (_Float16*)ws;                              // 2*B*D*N bytes
    float* si     = (float*)(ws + (size_t)2 * BB * DD * NN);    // B*N floats
    float* sj     = si + (size_t)BB * NN;                       // B*N floats
    float* sjmax  = sj + (size_t)BB * NN;                       // B floats (+pad)
    float2* FG    = (float2*)(sjmax + 4);                       // B*N float2

    gat_prep    <<<(BB * NN) / 256, 256, 0, stream>>>(x, emb, Ww, Wb, Aw, Ab, xpT, si, sj);
    gat_sjmax_fg<<<BB, 256, 0, stream>>>(sj, sjmax, FG);
    gat_attn    <<<BB * (NN / 16), 32, 0, stream>>>(msk, si, sjmax, FG, xpT, outp);
}
